// EncoderBlock_40561671143473
// MI455X (gfx1250) — matrix operand order repaired and checked by reading, not compiled, not run on hardware
//
#include <hip/hip_runtime.h>

typedef __attribute__((ext_vector_type(16))) _Float16 v16h;
typedef __attribute__((ext_vector_type(8)))  _Float16 v8h;
typedef __attribute__((ext_vector_type(8)))  float    v8f;

#define LN_EPS   1e-6f
#define NEG_BIG  (-1.0e9f)
#define FLT_BIG  (3.0e38f)

static __device__ inline v8h ld8h(const _Float16* p) { return *(const v8h*)p; }
static __device__ inline v16h mk16(v8h lo, v8h hi) {
  v16h r;
  #pragma unroll
  for (int i = 0; i < 8; ++i) { r[i] = lo[i]; r[8 + i] = hi[i]; }
  return r;
}

// --- CDNA5 async global->LDS DMA (ASYNCcnt-tracked), inline asm ------------
static __device__ inline void async_g2l_b128(unsigned lds_addr, const void* gptr) {
  asm volatile("global_load_async_to_lds_b128 %0, %1, off"
               :: "v"(lds_addr), "v"(gptr) : "memory");
}
static __device__ inline void wait_async0() {
  asm volatile("s_wait_asynccnt 0x0" ::: "memory");
}
static __device__ inline void wait_ds0() {
  asm volatile("s_wait_dscnt 0x0" ::: "memory");
}

// --- intra-16-lane xor butterfly via v_permlane16_b32 ----------------------
static __device__ inline float permxor16(float x, unsigned lo, unsigned hi, int mask) {
#if __has_builtin(__builtin_amdgcn_permlane16)
  unsigned u = __float_as_uint(x);
  return __uint_as_float(__builtin_amdgcn_permlane16(u, u, lo, hi, false, false));
#elif __has_builtin(__builtin_amdgcn_permlane16_b32)
  unsigned u = __float_as_uint(x);
  return __uint_as_float(__builtin_amdgcn_permlane16_b32(u, u, lo, hi, false, false));
#else
  return __shfl_xor(x, mask, 32);
#endif
}
static __device__ inline float rowmax16(float x) {
  x = fmaxf(x, permxor16(x, 0x67452301u, 0xEFCDAB89u, 1));
  x = fmaxf(x, permxor16(x, 0x54761032u, 0xDCFE98BAu, 2));
  x = fmaxf(x, permxor16(x, 0x32107654u, 0xBA98FEDCu, 4));
  x = fmaxf(x, permxor16(x, 0xFEDCBA98u, 0x76543210u, 8));
  return x;
}
static __device__ inline float rowsum16(float x) {
  x += permxor16(x, 0x67452301u, 0xEFCDAB89u, 1);
  x += permxor16(x, 0x54761032u, 0xDCFE98BAu, 2);
  x += permxor16(x, 0x32107654u, 0xBA98FEDCu, 4);
  x += permxor16(x, 0xFEDCBA98u, 0x76543210u, 8);
  return x;
}

// ---------------------------------------------------------------------------
// one-time tiny prep kernels
// ---------------------------------------------------------------------------
__global__ __launch_bounds__(256) void cvt_f32_f16(const float* __restrict__ in,
                                                   _Float16* __restrict__ out, int n) {
  int i = (blockIdx.x * 256 + threadIdx.x) * 4;
  if (i < n) {
    float4 v = *(const float4*)(in + i);
    out[i + 0] = (_Float16)v.x; out[i + 1] = (_Float16)v.y;
    out[i + 2] = (_Float16)v.z; out[i + 3] = (_Float16)v.w;
  }
}
__global__ __launch_bounds__(256) void cat3_512(const float* __restrict__ a,
                                                const float* __restrict__ b,
                                                const float* __restrict__ c,
                                                float* __restrict__ o) {
  int t = blockIdx.x * 256 + threadIdx.x;          // 0..1535 (grid 6)
  const float* s = t < 512 ? a : (t < 1024 ? b : c);
  o[t] = s[t & 511];
}

// ---------------------------------------------------------------------------
// LayerNorm over D=512 (unbiased variance, ddof=1), f16 out
// ---------------------------------------------------------------------------
__global__ __launch_bounds__(256) void ln512_f16(const float* __restrict__ x,
                                                 _Float16* __restrict__ y) {
  const int D = 512;
  size_t row = blockIdx.x;
  const float* xr = x + row * D;
  int t = threadIdx.x;
  float a0 = xr[t], a1 = xr[t + 256];
  float s = a0 + a1, q = a0 * a0 + a1 * a1;
  #pragma unroll
  for (int d = 1; d < 32; d <<= 1) { s += __shfl_xor(s, d, 32); q += __shfl_xor(q, d, 32); }
  __shared__ float rs[8], rq[8];
  if ((t & 31) == 0) { rs[t >> 5] = s; rq[t >> 5] = q; }
  __syncthreads();
  float S = 0.f, Q = 0.f;
  #pragma unroll
  for (int i = 0; i < 8; ++i) { S += rs[i]; Q += rq[i]; }
  float mu  = S * (1.0f / D);
  float var = (Q - S * mu) * (1.0f / (D - 1));
  float inv = rsqrtf(var + LN_EPS);
  _Float16* yr = y + row * D;
  yr[t]       = (_Float16)((a0 - mu) * inv);
  yr[t + 256] = (_Float16)((a1 - mu) * inv);
}

// ---------------------------------------------------------------------------
// Y = act(A(f16,MxK) @ W(f16,NxK)^T + bias) [+res], f32 and/or f16 out
// Block tile 128x128, BK=32, 8 waves x (32x64) = 2x4 WMMA each.
// Double-buffered LDS filled by async global->LDS DMA.
// ---------------------------------------------------------------------------
template<int RELU, int RES, int OUT32, int OUT16>
__global__ __launch_bounds__(256) void gemm_xwT(const _Float16* __restrict__ A,
                                                const _Float16* __restrict__ Wt,
                                                const float*    __restrict__ bias,
                                                const float*    __restrict__ res,
                                                float*          __restrict__ Yf,
                                                _Float16*       __restrict__ Yh,
                                                int M, int N, int K) {
  __shared__ _Float16 As[2][128 * 40];   // 128x32 tiles, row stride 40 halfs
  __shared__ _Float16 Bs[2][128 * 40];
  const int tid  = threadIdx.x;
  const int lane = tid & 31, wid = tid >> 5;
  const int hl = lane >> 4, lr = lane & 15;
  const int wm = wid & 3, wn = wid >> 2;          // 4x2 wave grid
  const int m0 = blockIdx.y * 128, n0 = blockIdx.x * 128;
  const int srow = tid >> 1, sc0 = (tid & 1) * 16;  // staging: 16 halfs/thread/tile

  v8f acc[2][4];
  #pragma unroll
  for (int a = 0; a < 2; ++a)
    #pragma unroll
    for (int b = 0; b < 4; ++b)
      #pragma unroll
      for (int j = 0; j < 8; ++j) acc[a][b][j] = 0.0f;

  const int nk = K >> 5;
  {
    const _Float16* ag = A  + (size_t)(m0 + srow) * K + sc0;
    const _Float16* wg = Wt + (size_t)(n0 + srow) * K + sc0;
    unsigned la = (unsigned)(unsigned long long)&As[0][srow * 40 + sc0];
    unsigned lb = (unsigned)(unsigned long long)&Bs[0][srow * 40 + sc0];
    async_g2l_b128(la, ag);       async_g2l_b128(la + 16, ag + 8);
    async_g2l_b128(lb, wg);       async_g2l_b128(lb + 16, wg + 8);
  }

  for (int kt = 0; kt < nk; ++kt) {
    const int cur = kt & 1;
    wait_async0();
    __syncthreads();
    if (kt + 1 < nk) {
      const int kk = (kt + 1) * 32;
      const _Float16* ag = A  + (size_t)(m0 + srow) * K + kk + sc0;
      const _Float16* wg = Wt + (size_t)(n0 + srow) * K + kk + sc0;
      unsigned la = (unsigned)(unsigned long long)&As[cur ^ 1][srow * 40 + sc0];
      unsigned lb = (unsigned)(unsigned long long)&Bs[cur ^ 1][srow * 40 + sc0];
      async_g2l_b128(la, ag);     async_g2l_b128(la + 16, ag + 8);
      async_g2l_b128(lb, wg);     async_g2l_b128(lb + 16, wg + 8);
    }

    v16h af[2], bf[4];
    #pragma unroll
    for (int mt = 0; mt < 2; ++mt) {
      const _Float16* p = &As[cur][(wm * 32 + mt * 16 + lr) * 40 + hl * 8];
      af[mt] = mk16(ld8h(p), ld8h(p + 16));
    }
    #pragma unroll
    for (int nt = 0; nt < 4; ++nt) {
      const _Float16* p = &Bs[cur][(wn * 64 + nt * 16 + lr) * 40 + hl * 8];
      bf[nt] = mk16(ld8h(p), ld8h(p + 16));
    }
    #pragma unroll
    for (int mt = 0; mt < 2; ++mt)
      #pragma unroll
      for (int nt = 0; nt < 4; ++nt)
        acc[mt][nt] = __builtin_amdgcn_wmma_f32_16x16x32_f16(
            false, af[mt], false, bf[nt], (short)0, acc[mt][nt], false, false);
    __syncthreads();
  }

  #pragma unroll
  for (int mt = 0; mt < 2; ++mt) {
    #pragma unroll
    for (int nt = 0; nt < 4; ++nt) {
      int col = n0 + wn * 64 + nt * 16 + lr;
      float bv = bias[col];
      #pragma unroll
      for (int i = 0; i < 8; ++i) {
        int row = m0 + wm * 32 + mt * 16 + hl * 8 + i;
        float v = acc[mt][nt][i] + bv;
        if (RELU) v = fmaxf(v, 0.0f);
        if (RES)  v += res[(size_t)row * N + col];
        if (OUT32) Yf[(size_t)row * N + col] = v;
        if (OUT16) Yh[(size_t)row * N + col] = (_Float16)v;
      }
    }
  }
}

// ---------------------------------------------------------------------------
// Flash attention: 4 waves x 16 q-rows per block; 32 keys/iter; d_k=64, H=8
// Q/K/V are views into the fused QKV buffer with row stride LD.
// ---------------------------------------------------------------------------
__global__ __launch_bounds__(128) void attn_fwd(const _Float16* __restrict__ Qm,
                                                const _Float16* __restrict__ Km,
                                                const _Float16* __restrict__ Vm,
                                                const int* __restrict__ mask,
                                                _Float16* __restrict__ O,
                                                int S, int LD) {
  const int DO = 512, DK = 64;
  __shared__ _Float16 Vt[64 * 40];        // V chunk transposed [d][key], pad 40
  __shared__ _Float16 Ps[4 * 16 * 40];    // per-wave P tile 16x32, pad 40
  const int tid  = threadIdx.x;
  const int lane = tid & 31, wave = tid >> 5;
  const int hl = lane >> 4, lr = lane & 15;
  const int b = blockIdx.y >> 3, h = blockIdx.y & 7;
  const int qbase = blockIdx.x * 64 + wave * 16;

  const _Float16* qrow = Qm + (size_t)(b * S + qbase + lr) * LD + h * DK + hl * 8;
  v16h qa0 = mk16(ld8h(qrow),      ld8h(qrow + 16));   // d 0..31
  v16h qa1 = mk16(ld8h(qrow + 32), ld8h(qrow + 48));   // d 32..63

  v8f o[4];
  float mrow[8], lrow[8];
  #pragma unroll
  for (int i = 0; i < 8; ++i) {
    mrow[i] = -FLT_BIG; lrow[i] = 0.0f;
    #pragma unroll
    for (int nb = 0; nb < 4; ++nb) o[nb][i] = 0.0f;
  }

  for (int kc = 0; kc < S; kc += 32) {
    { // stage V[kc..kc+31][0..63] transposed into LDS
      int r = tid >> 2, c0 = (tid & 3) * 16;
      const _Float16* vp = Vm + (size_t)(b * S + kc + r) * LD + h * DK + c0;
      v8h x0 = ld8h(vp), x1 = ld8h(vp + 8);
      #pragma unroll
      for (int e = 0; e < 8; ++e) {
        Vt[(c0 + e) * 40 + r]     = x0[e];
        Vt[(c0 + 8 + e) * 40 + r] = x1[e];
      }
    }
    __syncthreads();

    v8f sc[2];
    #pragma unroll
    for (int sub = 0; sub < 2; ++sub) {
      const _Float16* krow =
          Km + (size_t)(b * S + kc + sub * 16 + lr) * LD + h * DK + hl * 8;
      v16h kb0 = mk16(ld8h(krow),      ld8h(krow + 16));   // d 0..31
      v16h kb1 = mk16(ld8h(krow + 32), ld8h(krow + 48));   // d 32..63
      v8f c;
      #pragma unroll
      for (int j = 0; j < 8; ++j) c[j] = 0.0f;
      c = __builtin_amdgcn_wmma_f32_16x16x32_f16(false, qa0, false, kb0, (short)0, c, false, false);
      c = __builtin_amdgcn_wmma_f32_16x16x32_f16(false, qa1, false, kb1, (short)0, c, false, false);
      int mv = mask[b * S + kc + sub * 16 + lr];
      float mb = mv ? 0.0f : NEG_BIG;     // where(mask==0, -1e9, s/8) via fma
      #pragma unroll
      for (int j = 0; j < 8; ++j) sc[sub][j] = fmaf(c[j], 0.125f, mb);
    }

    // online softmax; row i lives across 16 lanes of acc element i
    #pragma unroll
    for (int i = 0; i < 8; ++i) {
      float t = rowmax16(fmaxf(sc[0][i], sc[1][i]));
      float mnew  = fmaxf(mrow[i], t);
      float alpha = __expf(mrow[i] - mnew);
      float p0 = __expf(sc[0][i] - mnew);
      float p1 = __expf(sc[1][i] - mnew);
      float rsum = rowsum16(p0 + p1);
      lrow[i] = lrow[i] * alpha + rsum;
      mrow[i] = mnew;
      #pragma unroll
      for (int nb = 0; nb < 4; ++nb) o[nb][i] *= alpha;
      _Float16* pr = &Ps[(wave * 16 + hl * 8 + i) * 40];   // C-layout -> LDS
      pr[lr]      = (_Float16)p0;
      pr[16 + lr] = (_Float16)p1;
    }
    wait_ds0();   // wave-local LDS RAW fence

    { // o += P(16x32) @ Vchunk(32x64)
      const _Float16* pp = &Ps[(wave * 16 + lr) * 40 + hl * 8];
      v16h pa = mk16(ld8h(pp), ld8h(pp + 16));
      #pragma unroll
      for (int nb = 0; nb < 4; ++nb) {
        const _Float16* vr = &Vt[(nb * 16 + lr) * 40 + hl * 8];
        v16h vb = mk16(ld8h(vr), ld8h(vr + 16));
        o[nb] = __builtin_amdgcn_wmma_f32_16x16x32_f16(
            false, pa, false, vb, (short)0, o[nb], false, false);
      }
    }
    __syncthreads();   // all waves done with Vt before restage
  }

  #pragma unroll
  for (int i = 0; i < 8; ++i) {
    float inv = 1.0f / lrow[i];
    size_t rbase = (size_t)(b * S + qbase + hl * 8 + i) * DO + h * DK;
    #pragma unroll
    for (int nb = 0; nb < 4; ++nb)
      O[rbase + nb * 16 + lr] = (_Float16)(o[nb][i] * inv);
  }
}

// ---------------------------------------------------------------------------
extern "C" void kernel_launch(void* const* d_in, const int* in_sizes, int n_in,
                              void* d_out, int out_size, void* d_ws, size_t ws_size,
                              hipStream_t stream) {
  const float* x    = (const float*)d_in[0];
  const int*   mask = (const int*)  d_in[1];
  const float* w_q = (const float*)d_in[2];   const float* b_q = (const float*)d_in[3];
  const float* w_k = (const float*)d_in[4];   const float* b_k = (const float*)d_in[5];
  const float* w_v = (const float*)d_in[6];   const float* b_v = (const float*)d_in[7];
  const float* w_o = (const float*)d_in[8];   const float* b_o = (const float*)d_in[9];
  const float* w_1 = (const float*)d_in[10];  const float* b_1 = (const float*)d_in[11];
  const float* w_2 = (const float*)d_in[12];  const float* b_2 = (const float*)d_in[13];

  const int B = 8, S = 2048, D = 512, F = 2048;
  const int M = B * S;                         // 16384 rows
  const size_t MB = 1024 * 1024;
  char* ws = (char*)d_ws;
  _Float16* xn   = (_Float16*)(ws +   0 * MB); // 16 MB (reused as attn out)
  _Float16* qkv  = (_Float16*)(ws +  16 * MB); // 48 MB fused QKV (M x 1536)
  float*    x1   = (float*)   (ws +  64 * MB); // 32 MB
  _Float16* h1   = (_Float16*)(ws +  96 * MB); // 64 MB
  _Float16* wq16 = (_Float16*)(ws + 160 * MB); // f16 weights (QKV contiguous!)
  _Float16* wo16 = wq16 +  786432;
  _Float16* w116 = wq16 + 1048576;             // 2048x512
  _Float16* w216 = w116 + 1048576;             // 512x2048
  float*    bqkv = (float*)(ws + 167 * MB);    // concatenated QKV bias (1536)
  _Float16* attn = xn;                         // reuse xn after QKV gemm
  _Float16* x1n  = qkv;                        // reuse qkv region after attention

  const int nDD = D * D, nFD = F * D;
  cvt_f32_f16<<<nDD / 1024, 256, 0, stream>>>(w_q, wq16,          nDD);
  cvt_f32_f16<<<nDD / 1024, 256, 0, stream>>>(w_k, wq16 + 262144, nDD);
  cvt_f32_f16<<<nDD / 1024, 256, 0, stream>>>(w_v, wq16 + 524288, nDD);
  cvt_f32_f16<<<nDD / 1024, 256, 0, stream>>>(w_o, wo16, nDD);
  cvt_f32_f16<<<nFD / 1024, 256, 0, stream>>>(w_1, w116, nFD);
  cvt_f32_f16<<<nFD / 1024, 256, 0, stream>>>(w_2, w216, nFD);
  cat3_512<<<6, 256, 0, stream>>>(b_q, b_k, b_v, bqkv);

  ln512_f16<<<M, 256, 0, stream>>>(x, xn);

  dim3 blk(256);
  dim3 gQKV(1536 / 128, M / 128);
  dim3 gD(D / 128, M / 128);
  dim3 gF(F / 128, M / 128);

  // fused QKV projection: (M x 512) @ (1536 x 512)^T -> (M x 1536) f16
  gemm_xwT<0,0,0,1><<<gQKV, blk, 0, stream>>>(xn, wq16, bqkv, nullptr, nullptr, qkv, M, 1536, D);

  attn_fwd<<<dim3(S / 64, B * 8), dim3(128), 0, stream>>>(
      qkv, qkv + 512, qkv + 1024, mask, attn, S, 1536);

  gemm_xwT<0,1,1,0><<<gD, blk, 0, stream>>>(attn, wo16, b_o, x, x1, nullptr, M, D, D);

  ln512_f16<<<M, 256, 0, stream>>>(x1, x1n);

  gemm_xwT<1,0,0,1><<<gF, blk, 0, stream>>>(x1n, w116, b_1, nullptr, nullptr, h1, M, F, D);
  gemm_xwT<0,1,1,0><<<gD, blk, 0, stream>>>(h1, w216, b_2, x1, (float*)d_out, nullptr, M, D, F);
}
